// NeuralOperator_21723944583763
// MI455X (gfx1250) — matrix operand order repaired and checked by reading, not compiled, not run on hardware
//
#include <hip/hip_runtime.h>

typedef __attribute__((ext_vector_type(16))) _Float16     v16h;
typedef __attribute__((ext_vector_type(8)))  float        v8f;
typedef __attribute__((ext_vector_type(8)))  unsigned int v8u;

#define NO_DEPTH 6
#define NO_WIDTH 64
#define NO_NS    512
#define NO_XS    1024
#define NO_NB    4
#define WPACK_DW (NO_DEPTH * 4 * 2 * 32 * 8) /* 12288 dwords = 48KB */

__device__ __forceinline__ float fast_tanh(float x) {
#if __has_builtin(__builtin_amdgcn_tanhf)
  return __builtin_amdgcn_tanhf(x);           // v_tanh_f32 (TRANS, co-executes with WMMA)
#else
  float ax = __builtin_fabsf(x);
  float e  = __builtin_amdgcn_exp2f(ax * -2.8853900817779268f);
  float r  = (1.0f - e) * __builtin_amdgcn_rcpf(1.0f + e);
  return __builtin_copysignf(r, x);
#endif
}

// Pack W_h (DEPTH,64,64) fp32 into f16 A-fragment order for Wt = W^T tiles.
// A 16x32 f16 order used here (contiguous k, same as B): lane L: M = L%16, hi = L/16;
//   VGPR v: K = 32*kc + 16*hi + 2*v + {0,1}
__global__ __launch_bounds__(256) void no_pack_weights(const float* __restrict__ W_h,
                                                       unsigned int* __restrict__ wpack) {
  int idx = blockIdx.x * 256 + threadIdx.x;
  if (idx >= WPACK_DW) return;
  int v    = idx & 7;
  int lane = (idx >> 3) & 31;
  int kc   = (idx >> 8) & 1;
  int t    = (idx >> 9) & 3;
  int l    = idx >> 11;
  int M  = t * 16 + (lane & 15);
  int hi = lane >> 4;
  int k  = 32 * kc + 16 * hi + 2 * v;
  // Wt[M][k] = W_h[l][k][M]
  float w0 = W_h[(l * NO_WIDTH + k) * NO_WIDTH + M];
  float w1 = W_h[(l * NO_WIDTH + k + 1) * NO_WIDTH + M];
  auto h2  = __builtin_amdgcn_cvt_pkrtz(w0, w1);
  wpack[idx] = __builtin_bit_cast(unsigned int, h2);
}

__global__ __launch_bounds__(256) void no_fused(
    const float* __restrict__ yu,   const float* __restrict__ x,
    const float* __restrict__ W_in, const float* __restrict__ b_in,
    const float* __restrict__ b_h,  const float* __restrict__ W_out,
    const float* __restrict__ b_out,
    const unsigned int* __restrict__ wpack, float* __restrict__ out) {

  __shared__ __align__(16) unsigned int wlds[WPACK_DW]; // 48KB packed A-fragments
  __shared__ float s_bh[NO_DEPTH * NO_WIDTH];
  __shared__ float s_win[NO_WIDTH];
  __shared__ float s_bin[NO_WIDTH];
  __shared__ float s_wout[NO_WIDTH];
  __shared__ float s_bout;
  __shared__ float s_x[32];
  __shared__ float s_acc[16 * 4];

  const int b   = blockIdx.y;
  const int xt0 = blockIdx.x * 16;
  const int tid = threadIdx.x;

  for (int i = tid; i < WPACK_DW; i += 256) wlds[i] = wpack[i];
  for (int i = tid; i < NO_DEPTH * NO_WIDTH; i += 256) s_bh[i] = b_h[i];
  if (tid < NO_WIDTH) { s_win[tid] = W_in[tid]; s_bin[tid] = b_in[tid]; s_wout[tid] = W_out[tid]; }
  if (tid == 0) s_bout = b_out[0];
  if (tid < 32) s_x[tid] = x[((size_t)b * NO_XS + xt0) * 2 + tid];
  if (tid < 64) s_acc[tid] = 0.0f;
  __syncthreads();

  const int wave = tid >> 5;
  const int lane = tid & 31;
  const int li   = lane & 15;   // column n (x-point within tile)
  const int hi   = lane >> 4;

  const float px0 = s_x[li * 2 + 0];
  const float px1 = s_x[li * 2 + 1];

  float acc0 = 0.f, acc1 = 0.f, acc2 = 0.f;

#pragma clang loop unroll(disable)
  for (int it = 0; it < NO_NS / 16; ++it) {
    const int sbase = it * 16 + wave * 2;

    float rr[2], u0[2], u1[2], u2[2];
#pragma unroll
    for (int s2 = 0; s2 < 2; ++s2) {
      const float* yus = yu + ((size_t)b * NO_NS + (sbase + s2)) * 5;
      u0[s2] = yus[0]; u1[s2] = yus[1]; u2[s2] = yus[2];
      float d0 = px0 - yus[3];
      float d1 = px1 - yus[4];
      rr[s2] = d0 * d0 + d1 * d1;
    }

    // h^T in D-fragment layout: d[t][s2] lane(li,hi) VGPR p -> feature m = 16t+8hi+p, col n=li
    v8f d[4][2];
#pragma unroll
    for (int t = 0; t < 4; ++t)
#pragma unroll
      for (int p = 0; p < 8; ++p) {
        int m = t * 16 + hi * 8 + p;
        float w = s_win[m], bi = s_bin[m];
        d[t][0][p] = rr[0] * w + bi;
        d[t][1][p] = rr[1] * w + bi;
      }

#pragma clang loop unroll(disable)
    for (int l = 0; l < NO_DEPTH; ++l) {
      const int wbase = l * (4 * 2 * 32 * 8);
      const int bbase = l * NO_WIDTH + hi * 8;

      // Build B fragments (h^T, f16, 32x16) from D fragments.
      v8u Bf[2][2];
#pragma unroll
      for (int s2 = 0; s2 < 2; ++s2) {
        unsigned int pk[4][4];
#pragma unroll
        for (int t = 0; t < 4; ++t)
#pragma unroll
          for (int q = 0; q < 4; ++q) {
            auto h2 = __builtin_amdgcn_cvt_pkrtz(d[t][s2][2 * q], d[t][s2][2 * q + 1]);
            pk[t][q] = __builtin_bit_cast(unsigned int, h2);
          }
#if __has_builtin(__builtin_amdgcn_permlane16_swap)
        // One v_permlane16_swap_b32 produces {X.lo,Y.lo} and {X.hi,Y.hi}:
        // exactly Bf[kc][q] and Bf[kc][4+q].
#pragma unroll
        for (int kc = 0; kc < 2; ++kc)
#pragma unroll
          for (int q = 0; q < 4; ++q) {
            auto r = __builtin_amdgcn_permlane16_swap(pk[2 * kc][q], pk[2 * kc + 1][q],
                                                      false, false);
            Bf[kc][s2][q]     = r[0];
            Bf[kc][s2][4 + q] = r[1];
          }
#else
        unsigned int xp[4][4];
#pragma unroll
        for (int t = 0; t < 4; ++t)
#pragma unroll
          for (int q = 0; q < 4; ++q)
            xp[t][q] = (unsigned int)__shfl_xor((int)pk[t][q], 16, 32);
#pragma unroll
        for (int kc = 0; kc < 2; ++kc)
#pragma unroll
          for (int q = 0; q < 4; ++q) {
            Bf[kc][s2][q]     = hi ? xp[2 * kc + 1][q] : pk[2 * kc][q];
            Bf[kc][s2][4 + q] = hi ? pk[2 * kc + 1][q] : xp[2 * kc][q];
          }
#endif
      }

      // B is fully built from d; process one m-tile at a time so only one nd
      // pair (16 VGPRs) is live beyond d/Bf.
#pragma unroll
      for (int t = 0; t < 4; ++t) {
        v8f c;
#pragma unroll
        for (int p = 0; p < 8; ++p) c[p] = s_bh[bbase + t * 16 + p];
        v8f nd0 = c, nd1 = c;
#pragma unroll
        for (int kc = 0; kc < 2; ++kc) {
          const v8u* ap = (const v8u*)&wlds[wbase + ((t * 2 + kc) * 32 + lane) * 8];
          v16h A  = __builtin_bit_cast(v16h, *ap);
          v16h B0 = __builtin_bit_cast(v16h, Bf[kc][0]);
          v16h B1 = __builtin_bit_cast(v16h, Bf[kc][1]);
          nd0 = __builtin_amdgcn_wmma_f32_16x16x32_f16(false, A, false, B0, (short)0, nd0, false, false);
          nd1 = __builtin_amdgcn_wmma_f32_16x16x32_f16(false, A, false, B1, (short)0, nd1, false, false);
        }
#pragma unroll
        for (int p = 0; p < 8; ++p) {
          d[t][0][p] = fast_tanh(nd0[p]) + d[t][0][p];
          d[t][1][p] = fast_tanh(nd1[p]) + d[t][1][p];
        }
      }
    }

    // Output layer: k[n] = sum_m h[m][n]*W_out[m] + b_out, then channel accumulation.
#pragma unroll
    for (int s2 = 0; s2 < 2; ++s2) {
      float part = 0.f;
#pragma unroll
      for (int t = 0; t < 4; ++t)
#pragma unroll
        for (int p = 0; p < 8; ++p)
          part += d[t][s2][p] * s_wout[t * 16 + hi * 8 + p];
      part += __shfl_xor(part, 16, 32);
      float k = part + s_bout;
      acc0 += k * u0[s2];
      acc1 += k * u1[s2];
      acc2 += k * u2[s2];
    }
  }

  if (hi == 0) { // hi half holds duplicates
    atomicAdd(&s_acc[li * 4 + 0], acc0);
    atomicAdd(&s_acc[li * 4 + 1], acc1);
    atomicAdd(&s_acc[li * 4 + 2], acc2);
  }
  __syncthreads();
  if (tid < 16) {
    const float inv = 1.0f / (float)NO_NS;
    size_t o = ((size_t)b * NO_XS + xt0 + tid) * 3;
    out[o + 0] = s_acc[tid * 4 + 0] * inv;
    out[o + 1] = s_acc[tid * 4 + 1] * inv;
    out[o + 2] = s_acc[tid * 4 + 2] * inv;
  }
}

extern "C" void kernel_launch(void* const* d_in, const int* in_sizes, int n_in,
                              void* d_out, int out_size, void* d_ws, size_t ws_size,
                              hipStream_t stream) {
  const float* yu    = (const float*)d_in[0];
  const float* x     = (const float*)d_in[1];
  const float* W_in  = (const float*)d_in[2];
  const float* b_in  = (const float*)d_in[3];
  const float* W_h   = (const float*)d_in[4];
  const float* b_h   = (const float*)d_in[5];
  const float* W_out = (const float*)d_in[6];
  const float* b_out = (const float*)d_in[7];
  unsigned int* wpack = (unsigned int*)d_ws;   // needs 48KB of scratch
  float* out = (float*)d_out;

  no_pack_weights<<<(WPACK_DW + 255) / 256, 256, 0, stream>>>(W_h, wpack);
  dim3 grid(NO_XS / 16, NO_NB);
  no_fused<<<grid, 256, 0, stream>>>(yu, x, W_in, b_in, b_h, W_out, b_out, wpack, out);
}
